// EdgeAwareGATCoxModel_28106265985392
// MI455X (gfx1250) — matrix operand order repaired and checked by reading, not compiled, not run on hardware
//
#include <hip/hip_runtime.h>
#include <hip/hip_bf16.h>
#include <float.h>

// ---------------- model dimensions ----------------
#define NN 20000
#define EE 320000
#define FF 200
#define HID 96
#define HEADS 4
#define EHID 8
#define CLIN 32
#define MET 128
#define BB 64
#define HC1 (HEADS * HID)   // 384
#define FIN (3 * HID + CLIN + MET) // 448

// ---------------- WMMA types ----------------
typedef __attribute__((ext_vector_type(16))) __bf16 v16bf;
typedef __attribute__((ext_vector_type(8)))  float  v8f;

__device__ __forceinline__ unsigned short f2bf_bits(float f) {
    union { float f; unsigned u; } v; v.f = f;
    unsigned r = v.u + 0x7FFFu + ((v.u >> 16) & 1u);  // round-to-nearest-even
    return (unsigned short)(r >> 16);
}

__device__ __forceinline__ void atomicMaxFloat(float* addr, float val) {
    if (val >= 0.0f) atomicMax((int*)addr, __float_as_int(val));
    else             atomicMin((unsigned int*)addr, __float_as_uint(val));
}

// ---------------- generic fill ----------------
__global__ void k_fill(float* __restrict__ p, float v, long n) {
    long i = (long)blockIdx.x * blockDim.x + threadIdx.x;
    if (i < n) p[i] = v;
}

// ---------------- fragment packing ----------------
// A fragment layout (16x32 bf16 tile, wave32): lane = (half<<4)|r where r = M%16,
// element e -> K-offset ko = 16*(e>>3) + 8*half + (e&7).
// Packed so each lane's 16 elements are contiguous (32B): one b128-pair load.
__global__ void k_packA(const float* __restrict__ A, unsigned short* __restrict__ Ap,
                        int M, int K, int Kpad) {
    long t = (long)blockIdx.x * blockDim.x + threadIdx.x;
    long tot = (long)M * Kpad;
    if (t >= tot) return;
    int m  = (int)(t / Kpad);
    int kk = (int)(t % Kpad);
    int tm = m >> 4, r = m & 15;
    int kc = kk >> 5, ko = kk & 31;
    int half = (ko >> 3) & 1;
    int e = ((ko >> 4) << 3) | (ko & 7);
    int lane = (half << 4) | r;
    long dst = ((((long)tm * (Kpad >> 5) + kc) * 32 + lane) << 4) + e;
    float v = (kk < K) ? A[(long)m * K + kk] : 0.0f;
    Ap[dst] = f2bf_bits(v);
}

// B fragment layout (32x16 bf16 tile): lane = (half<<4)|r where r = N%16,
// element e -> K-offset ko = 16*(e>>3) + 8*half + (e&7).
__global__ void k_packB(const float* __restrict__ B, unsigned short* __restrict__ Bp,
                        int K, int Nc, int Kpad) {
    long t = (long)blockIdx.x * blockDim.x + threadIdx.x;
    long tot = (long)Kpad * Nc;
    if (t >= tot) return;
    int kk  = (int)(t / Nc);
    int col = (int)(t % Nc);
    int tn = col >> 4, r = col & 15;
    int kc = kk >> 5, ko = kk & 31;
    int half = (ko >> 3) & 1;
    int e = ((ko >> 4) << 3) | (ko & 7);
    int lane = (half << 4) | r;
    long dst = ((((long)tn * (Kpad >> 5) + kc) * 32 + lane) << 4) + e;
    float v = (kk < K) ? B[(long)kk * Nc + col] : 0.0f;
    Bp[dst] = f2bf_bits(v);
}

// ---------------- WMMA bf16 GEMM on pre-packed fragments ----------------
// C[MxNc] = A[MxK] @ B[KxNc]; M,Nc multiples of 16; K padded to 32*KC.
// epilogue: 0 = none, 1 = +bias then relu, 2 = +bias
__global__ void k_gemm_wmma(const unsigned short* __restrict__ Ap,
                            const unsigned short* __restrict__ Bp,
                            float* __restrict__ C, const float* __restrict__ bias,
                            int M, int Nc, int KC, int epilogue) {
    int wave = blockIdx.x * (blockDim.x >> 5) + (threadIdx.x >> 5);
    int lane = threadIdx.x & 31;
    int tilesN = Nc >> 4;
    int tiles = (M >> 4) * tilesN;
    if (wave >= tiles) return;                 // wave-uniform exit; EXEC full inside
    int tm = wave / tilesN, tn = wave % tilesN;
    const v16bf* Abase = (const v16bf*)Ap + (long)tm * KC * 32 + lane;
    const v16bf* Bbase = (const v16bf*)Bp + (long)tn * KC * 32 + lane;
    v8f acc = {};
    for (int kc = 0; kc < KC; ++kc) {
        v16bf a = Abase[(long)kc * 32];        // 32B contiguous per lane -> b128 pair
        v16bf b = Bbase[(long)kc * 32];
        __builtin_prefetch((const void*)(Abase + (long)(kc + 1) * 32), 0, 1);
        __builtin_prefetch((const void*)(Bbase + (long)(kc + 1) * 32), 0, 1);
        acc = __builtin_amdgcn_wmma_f32_16x16x32_bf16(false, a, false, b,
                                                      (short)0, acc, false, false);
    }
    // C 16x16 f32 layout: VGPR i -> M = 16*tm + i + 8*half ; N = 16*tn + (lane&15)
    int r = lane & 15, half = lane >> 4;
    int col = (tn << 4) + r;
    float bcol = epilogue ? bias[col] : 0.0f;
#pragma unroll
    for (int i = 0; i < 8; ++i) {
        int rr = (tm << 4) + i + (half << 3);
        float v = acc[i] + bcol;
        if (epilogue == 1) v = fmaxf(v, 0.0f);
        C[(long)rr * Nc + col] = v;
    }
}

// ---------------- edge MLP + self-loop edge-attr accumulation ----------------
__global__ void k_edge_mlp(const float* __restrict__ ea_in, const int* __restrict__ dst,
                           const float* __restrict__ w1, const float* __restrict__ b1,
                           const float* __restrict__ w2, const float* __restrict__ b2,
                           float* __restrict__ ea, float* __restrict__ loopacc,
                           float* __restrict__ cnt) {
    int e = blockIdx.x * blockDim.x + threadIdx.x;
    if (e >= EE) return;
    float v = ea_in[e];
    float h[EHID], o[EHID];
#pragma unroll
    for (int j = 0; j < EHID; ++j) h[j] = fmaxf(v * w1[j] + b1[j], 0.0f);
#pragma unroll
    for (int j = 0; j < EHID; ++j) {
        float s = b2[j];
#pragma unroll
        for (int i = 0; i < EHID; ++i) s += h[i] * w2[i * EHID + j];
        o[j] = fmaxf(s, 0.0f);
        ea[(long)e * EHID + j] = o[j];
    }
    int d = dst[e];
#pragma unroll
    for (int j = 0; j < EHID; ++j) atomicAdd(&loopacc[(long)d * EHID + j], o[j]);
    atomicAdd(&cnt[d], 1.0f);
}

__global__ void k_loop_div(const float* __restrict__ loopacc, const float* __restrict__ cnt,
                           float* __restrict__ loopea) {
    int n = blockIdx.x * blockDim.x + threadIdx.x;
    if (n >= NN) return;
    float c = fmaxf(cnt[n], 1.0f);
#pragma unroll
    for (int j = 0; j < EHID; ++j) loopea[(long)n * EHID + j] = loopacc[(long)n * EHID + j] / c;
}

// ---------------- per-node attention terms ----------------
__global__ void k_attn_node(const float* __restrict__ hmat, const float* __restrict__ a_s,
                            const float* __restrict__ a_d, float* __restrict__ asrc,
                            float* __restrict__ adst, int H, int C) {
    int t = blockIdx.x * blockDim.x + threadIdx.x;
    if (t >= NN * H) return;
    int n = t / H, h = t % H;
    const float* row = &hmat[(long)n * H * C + (long)h * C];
    float ss = 0.0f, sd = 0.0f;
    for (int c = 0; c < C; ++c) { float v = row[c]; ss += v * a_s[h * C + c]; sd += v * a_d[h * C + c]; }
    asrc[t] = ss; adst[t] = sd;
}

// ae_proj[j,h] = sum_c lin_e[j, h*C + c] * a_e[h, c]
__global__ void k_aeproj(const float* __restrict__ lin_e, const float* __restrict__ a_e,
                         float* __restrict__ aep, int H, int C) {
    int t = threadIdx.x;
    if (t >= EHID * H) return;
    int j = t / H, h = t % H;
    float s = 0.0f;
    for (int c = 0; c < C; ++c) s += lin_e[(long)j * H * C + h * C + c] * a_e[h * C + c];
    aep[j * H + h] = s;
}

// ---------------- edge logits + segment max ----------------
__global__ void k_logits(const int* __restrict__ src, const int* __restrict__ dst,
                         const float* __restrict__ ea, const float* __restrict__ loopea,
                         const float* __restrict__ aep, const float* __restrict__ asrc,
                         const float* __restrict__ adst, float* __restrict__ logits,
                         float* __restrict__ mbuf, int H) {
    int t = blockIdx.x * blockDim.x + threadIdx.x;
    int tot = (EE + NN) * H;
    if (t >= tot) return;
    int ep = t / H, h = t % H;
    int s, d; const float* ev;
    if (ep < EE) { s = src[ep]; d = dst[ep]; ev = &ea[(long)ep * EHID]; }
    else { s = d = ep - EE; ev = &loopea[(long)s * EHID]; }
    float ae = 0.0f;
#pragma unroll
    for (int j = 0; j < EHID; ++j) ae += ev[j] * aep[j * H + h];
    float l = asrc[s * H + h] + adst[d * H + h] + ae;
    l = (l > 0.0f) ? l : 0.2f * l;                 // leaky_relu(0.2)
    logits[(long)ep * H + h] = l;
    atomicMaxFloat(&mbuf[d * H + h], l);
}

// ---------------- exp + segment sum (in-place on logits) ----------------
__global__ void k_expw(const int* __restrict__ dst, float* __restrict__ logits,
                       const float* __restrict__ mbuf, float* __restrict__ den, int H) {
    int t = blockIdx.x * blockDim.x + threadIdx.x;
    int tot = (EE + NN) * H;
    if (t >= tot) return;
    int ep = t / H, h = t % H;
    int d = (ep < EE) ? dst[ep] : (ep - EE);
    float w = expf(logits[(long)ep * H + h] - mbuf[d * H + h]);
    logits[(long)ep * H + h] = w;
    atomicAdd(&den[d * H + h], w);
}

// ---------------- weighted scatter aggregation (8 channels / thread) ----------------
__global__ void k_agg(const int* __restrict__ src, const int* __restrict__ dst,
                      const float* __restrict__ wbuf, const float* __restrict__ den,
                      const float* __restrict__ hmat, float* __restrict__ out,
                      int H, int C) {
    int groups = C / 8;
    long t = (long)blockIdx.x * blockDim.x + threadIdx.x;
    long tot = (long)(EE + NN) * H * groups;
    if (t >= tot) return;
    int g = (int)(t % groups);
    long rem = t / groups;
    int h = (int)(rem % H);
    int ep = (int)(rem / H);
    int s, d;
    if (ep < EE) { s = src[ep]; d = dst[ep]; } else { s = d = ep - EE; }
    float w = wbuf[(long)ep * H + h] / den[d * H + h];
    const float* hp = &hmat[(long)s * H * C + (long)h * C + g * 8];
    float* op = &out[(long)d * H * C + (long)h * C + g * 8];
#pragma unroll
    for (int c = 0; c < 8; ++c) atomicAdd(&op[c], w * hp[c]);
}

// ---------------- LayerNorm + ReLU (one wave per node) ----------------
__global__ void k_ln_relu(const float* __restrict__ in, const float* __restrict__ bias,
                          const float* __restrict__ extra, const float* __restrict__ g,
                          const float* __restrict__ b, float* __restrict__ out, int D) {
    int wid = blockIdx.x * (blockDim.x >> 5) + (threadIdx.x >> 5);
    int lane = threadIdx.x & 31;
    if (wid >= NN) return;
    const float* row = &in[(long)wid * D];
    const float* ex  = extra ? &extra[(long)wid * D] : nullptr;
    float s = 0.0f, s2 = 0.0f;
    for (int d = lane; d < D; d += 32) {
        float v = row[d] + bias[d] + (ex ? ex[d] : 0.0f);
        s += v; s2 += v * v;
    }
    for (int o = 16; o > 0; o >>= 1) { s += __shfl_xor(s, o, 32); s2 += __shfl_xor(s2, o, 32); }
    float mu  = s / D;
    float var = s2 / D - mu * mu;
    float inv = rsqrtf(var + 1e-5f);
    for (int d = lane; d < D; d += 32) {
        float v = row[d] + bias[d] + (ex ? ex[d] : 0.0f);
        out[(long)wid * D + d] = fmaxf((v - mu) * inv * g[d] + b[d], 0.0f);
    }
}

// ---------------- gate MLP ----------------
__global__ void k_gate_hid(const float* __restrict__ x2, const float* __restrict__ w1,
                           const float* __restrict__ b1, float* __restrict__ ghid) {
    int t = blockIdx.x * blockDim.x + threadIdx.x;
    if (t >= NN * (HID / 2)) return;
    int n = t / (HID / 2), j = t % (HID / 2);
    float s = b1[j];
    const float* row = &x2[(long)n * HID];
    for (int c = 0; c < HID; ++c) s += row[c] * w1[c * (HID / 2) + j];
    ghid[t] = fmaxf(s, 0.0f);
}

__global__ void k_gate(const float* __restrict__ ghid, const float* __restrict__ w2,
                       const float* __restrict__ b2, const int* __restrict__ bv,
                       float* __restrict__ gate, float* __restrict__ gm,
                       float* __restrict__ gcnt) {
    int n = blockIdx.x * blockDim.x + threadIdx.x;
    if (n >= NN) return;
    float s = b2[0];
    const float* row = &ghid[(long)n * (HID / 2)];
#pragma unroll
    for (int j = 0; j < HID / 2; ++j) s += row[j] * w2[j];
    gate[n] = s;
    int b = bv[n];
    atomicAdd(&gcnt[b], 1.0f);
    atomicMaxFloat(&gm[b], s);
}

__global__ void k_gate_exp(const int* __restrict__ bv, float* __restrict__ gate,
                           const float* __restrict__ gm, float* __restrict__ gden) {
    int n = blockIdx.x * blockDim.x + threadIdx.x;
    if (n >= NN) return;
    int b = bv[n];
    float ge = expf(gate[n] - gm[b]);
    gate[n] = ge;
    atomicAdd(&gden[b], ge);
}

// ---------------- pooling ----------------
__global__ void k_pool1(const float* __restrict__ x2, const int* __restrict__ bv,
                        float* __restrict__ meanacc, float* __restrict__ maxacc) {
    int t = blockIdx.x * blockDim.x + threadIdx.x;
    if (t >= NN * HID) return;
    int n = t / HID, d = t % HID;
    int b = bv[n];
    float v = x2[(long)n * HID + d];
    atomicAdd(&meanacc[b * HID + d], v);
    atomicMaxFloat(&maxacc[b * HID + d], v);
}

__global__ void k_pool2(const float* __restrict__ x2, const int* __restrict__ bv,
                        const float* __restrict__ ge, const float* __restrict__ gden,
                        float* __restrict__ attacc) {
    int t = blockIdx.x * blockDim.x + threadIdx.x;
    if (t >= NN * HID) return;
    int n = t / HID, d = t % HID;
    int b = bv[n];
    atomicAdd(&attacc[b * HID + d], (ge[n] / gden[b]) * x2[(long)n * HID + d]);
}

// ---------------- fusion head (one block per graph) ----------------
__global__ void k_head(const float* __restrict__ meanacc, const float* __restrict__ maxacc,
                       const float* __restrict__ attacc, const float* __restrict__ gcnt,
                       const float* __restrict__ clin, const float* __restrict__ met,
                       const float* __restrict__ fw1, const float* __restrict__ fb1,
                       const float* __restrict__ fw2, const float* __restrict__ fb2,
                       const float* __restrict__ rw, const float* __restrict__ rb,
                       float* __restrict__ outRisk, float* __restrict__ outGE,
                       float* __restrict__ outLat) {
    __shared__ float fused[FIN];
    __shared__ float lat1[HID];
    __shared__ float lat2[HID / 2];
    int b = blockIdx.x, t = threadIdx.x;
    for (int i = t; i < FIN; i += blockDim.x) {
        float v;
        if (i < HID)            v = meanacc[b * HID + i] / gcnt[b];
        else if (i < 2 * HID)   v = maxacc[b * HID + (i - HID)];
        else if (i < 3 * HID)   v = attacc[b * HID + (i - 2 * HID)];
        else if (i < 3 * HID + CLIN) v = clin[b * CLIN + (i - 3 * HID)];
        else                    v = met[b * MET + (i - 3 * HID - CLIN)];
        fused[i] = v;
        if (i < 3 * HID) outGE[(long)b * 3 * HID + i] = v;
    }
    __syncthreads();
    for (int j = t; j < HID; j += blockDim.x) {
        float s = fb1[j];
        for (int i = 0; i < FIN; ++i) s += fused[i] * fw1[i * HID + j];
        lat1[j] = fmaxf(s, 0.0f);
    }
    __syncthreads();
    for (int j = t; j < HID / 2; j += blockDim.x) {
        float s = fb2[j];
        for (int i = 0; i < HID; ++i) s += lat1[i] * fw2[i * (HID / 2) + j];
        lat2[j] = fmaxf(s, 0.0f);
        outLat[(long)b * (HID / 2) + j] = lat2[j];
    }
    __syncthreads();
    if (t == 0) {
        float s = rb[0];
        for (int i = 0; i < HID / 2; ++i) s += lat2[i] * rw[i];
        outRisk[b] = s;
    }
}

// ==================================================================
extern "C" void kernel_launch(void* const* d_in, const int* in_sizes, int n_in,
                              void* d_out, int out_size, void* d_ws, size_t ws_size,
                              hipStream_t stream) {
    const float* x        = (const float*)d_in[0];
    const int*   ei       = (const int*)d_in[1];
    const float* edge_attr= (const float*)d_in[2];
    const int*   bv       = (const int*)d_in[3];
    const float* clinical = (const float*)d_in[4];
    const float* metab    = (const float*)d_in[5];
    const float* np_w = (const float*)d_in[6],  *np_b = (const float*)d_in[7];
    const float* ee_w1= (const float*)d_in[8],  *ee_b1= (const float*)d_in[9];
    const float* ee_w2= (const float*)d_in[10], *ee_b2= (const float*)d_in[11];
    const float* g1_lin=(const float*)d_in[12], *g1_as=(const float*)d_in[13];
    const float* g1_ad= (const float*)d_in[14], *g1_le=(const float*)d_in[15];
    const float* g1_ae= (const float*)d_in[16], *g1_b =(const float*)d_in[17];
    const float* g2_lin=(const float*)d_in[18], *g2_as=(const float*)d_in[19];
    const float* g2_ad= (const float*)d_in[20], *g2_le=(const float*)d_in[21];
    const float* g2_ae= (const float*)d_in[22], *g2_b =(const float*)d_in[23];
    const float* res_w= (const float*)d_in[24], *res_b=(const float*)d_in[25];
    const float* ln1_g= (const float*)d_in[26], *ln1_b=(const float*)d_in[27];
    const float* ln2_g= (const float*)d_in[28], *ln2_b=(const float*)d_in[29];
    const float* gp_w1= (const float*)d_in[30], *gp_b1=(const float*)d_in[31];
    const float* gp_w2= (const float*)d_in[32], *gp_b2=(const float*)d_in[33];
    const float* f_w1 = (const float*)d_in[34], *f_b1 =(const float*)d_in[35];
    const float* f_w2 = (const float*)d_in[36], *f_b2 =(const float*)d_in[37];
    const float* r_w  = (const float*)d_in[38], *r_b  =(const float*)d_in[39];

    const int* src = ei;
    const int* dst = ei + EE;

    // ---- workspace arena (floats, 64B-aligned slots) ----
    float* W = (float*)d_ws;
    size_t off = 0;
    auto alloc = [&](size_t n) { float* p = W + off; off += (n + 15) & ~(size_t)15; return p; };
    float* ea      = alloc((size_t)EE * EHID);
    float* loopacc = alloc((size_t)NN * EHID);
    float* loopea  = alloc((size_t)NN * EHID);
    float* cnt     = alloc(NN);
    float* x0      = alloc((size_t)NN * HID);
    float* h1      = alloc((size_t)NN * HC1);   // later reused as x1
    float* out1    = alloc((size_t)NN * HC1);
    float* asrc1   = alloc((size_t)NN * HEADS);
    float* adst1   = alloc((size_t)NN * HEADS);
    float* aep1    = alloc(64);
    float* logit1  = alloc((size_t)(EE + NN) * HEADS);
    float* m1      = alloc((size_t)NN * HEADS);
    float* den1    = alloc((size_t)NN * HEADS);
    float* h2      = alloc((size_t)NN * HID);
    float* asrc2   = alloc(NN);
    float* adst2   = alloc(NN);
    float* aep2    = alloc(16);
    float* logit2  = alloc(EE + NN);
    float* m2      = alloc(NN);
    float* den2    = alloc(NN);
    float* out2    = alloc((size_t)NN * HID);
    float* xres    = alloc((size_t)NN * HID);
    float* x2      = alloc((size_t)NN * HID);
    float* ghid    = alloc((size_t)NN * (HID / 2));
    float* gate    = alloc(NN);
    float* gm      = alloc(BB);
    float* gden    = alloc(BB);
    float* gcnt    = alloc(BB);
    float* meanacc = alloc(BB * HID);
    float* maxacc  = alloc(BB * HID);
    float* attacc  = alloc(BB * HID);
    // bf16 fragment-packed staging (ushort): max A = NN x 384, max B = 384x96 padded
    unsigned short* Apack = (unsigned short*)alloc(((size_t)NN * HC1) / 2 + 16);
    unsigned short* Bpack = (unsigned short*)alloc(36864 / 2 + 16);

    float* outRisk = (float*)d_out;
    float* outGE   = outRisk + BB;
    float* outLat  = outGE + (size_t)BB * 3 * HID;

    auto fill = [&](float* p, float v, long n) {
        long g = (n + 255) / 256;
        k_fill<<<dim3((unsigned)g), dim3(256), 0, stream>>>(p, v, n);
    };
    auto gemm = [&](const float* A, const float* Bm, float* C, const float* bias,
                    int M, int Nc, int K, int ep) {
        int Kpad = (K + 31) & ~31;
        int KC = Kpad >> 5;
        long atot = (long)M * Kpad;
        k_packA<<<dim3((unsigned)((atot + 255) / 256)), dim3(256), 0, stream>>>(A, Apack, M, K, Kpad);
        long btot = (long)Kpad * Nc;
        k_packB<<<dim3((unsigned)((btot + 255) / 256)), dim3(256), 0, stream>>>(Bm, Bpack, K, Nc, Kpad);
        int tiles = (M >> 4) * (Nc >> 4);
        k_gemm_wmma<<<dim3((tiles + 3) / 4), dim3(128), 0, stream>>>(Apack, Bpack, C, bias, M, Nc, KC, ep);
    };

    // ---- edge encoder + self-loop edge attrs ----
    fill(cnt, 0.0f, NN);
    fill(loopacc, 0.0f, (long)NN * EHID);
    k_edge_mlp<<<dim3((EE + 255) / 256), dim3(256), 0, stream>>>(
        edge_attr, dst, ee_w1, ee_b1, ee_w2, ee_b2, ea, loopacc, cnt);
    k_loop_div<<<dim3((NN + 255) / 256), dim3(256), 0, stream>>>(loopacc, cnt, loopea);

    // ---- node projection (WMMA) ----
    gemm(x, np_w, x0, np_b, NN, HID, FF, 1);

    // ---- GAT layer 1 ----
    gemm(x0, g1_lin, h1, nullptr, NN, HC1, HID, 0);
    k_attn_node<<<dim3((NN * HEADS + 255) / 256), dim3(256), 0, stream>>>(
        h1, g1_as, g1_ad, asrc1, adst1, HEADS, HID);
    k_aeproj<<<dim3(1), dim3(64), 0, stream>>>(g1_le, g1_ae, aep1, HEADS, HID);
    fill(m1, -FLT_MAX, (long)NN * HEADS);
    fill(den1, 0.0f, (long)NN * HEADS);
    fill(out1, 0.0f, (long)NN * HC1);
    {
        int tot = (EE + NN) * HEADS;
        k_logits<<<dim3((tot + 255) / 256), dim3(256), 0, stream>>>(
            src, dst, ea, loopea, aep1, asrc1, adst1, logit1, m1, HEADS);
        k_expw<<<dim3((tot + 255) / 256), dim3(256), 0, stream>>>(dst, logit1, m1, den1, HEADS);
        long atot = (long)(EE + NN) * HEADS * (HID / 8);
        k_agg<<<dim3((unsigned)((atot + 255) / 256)), dim3(256), 0, stream>>>(
            src, dst, logit1, den1, h1, out1, HEADS, HID);
    }
    // x1 = relu(LN(out1 + g1_b))  -> write into h1
    k_ln_relu<<<dim3((NN + 7) / 8), dim3(256), 0, stream>>>(out1, g1_b, nullptr, ln1_g, ln1_b, h1, HC1);

    // ---- GAT layer 2 + residual path ----
    gemm(h1, g2_lin, h2, nullptr, NN, HID, HC1, 0);
    gemm(x0, res_w, xres, res_b, NN, HID, HID, 2);
    k_attn_node<<<dim3((NN + 255) / 256), dim3(256), 0, stream>>>(
        h2, g2_as, g2_ad, asrc2, adst2, 1, HID);
    k_aeproj<<<dim3(1), dim3(64), 0, stream>>>(g2_le, g2_ae, aep2, 1, HID);
    fill(m2, -FLT_MAX, NN);
    fill(den2, 0.0f, NN);
    fill(out2, 0.0f, (long)NN * HID);
    {
        int tot = EE + NN;
        k_logits<<<dim3((tot + 255) / 256), dim3(256), 0, stream>>>(
            src, dst, ea, loopea, aep2, asrc2, adst2, logit2, m2, 1);
        k_expw<<<dim3((tot + 255) / 256), dim3(256), 0, stream>>>(dst, logit2, m2, den2, 1);
        long atot = (long)(EE + NN) * (HID / 8);
        k_agg<<<dim3((unsigned)((atot + 255) / 256)), dim3(256), 0, stream>>>(
            src, dst, logit2, den2, h2, out2, 1, HID);
    }
    // x2 = relu(LN(out2 + g2_b + xres))
    k_ln_relu<<<dim3((NN + 7) / 8), dim3(256), 0, stream>>>(out2, g2_b, xres, ln2_g, ln2_b, x2, HID);

    // ---- pooling ----
    fill(gm, -FLT_MAX, BB);
    fill(gden, 0.0f, BB);
    fill(gcnt, 0.0f, BB);
    fill(meanacc, 0.0f, BB * HID);
    fill(maxacc, -FLT_MAX, BB * HID);
    fill(attacc, 0.0f, BB * HID);
    k_gate_hid<<<dim3((NN * (HID / 2) + 255) / 256), dim3(256), 0, stream>>>(x2, gp_w1, gp_b1, ghid);
    k_gate<<<dim3((NN + 255) / 256), dim3(256), 0, stream>>>(ghid, gp_w2, gp_b2, bv, gate, gm, gcnt);
    k_pool1<<<dim3((NN * HID + 255) / 256), dim3(256), 0, stream>>>(x2, bv, meanacc, maxacc);
    k_gate_exp<<<dim3((NN + 255) / 256), dim3(256), 0, stream>>>(bv, gate, gm, gden);
    k_pool2<<<dim3((NN * HID + 255) / 256), dim3(256), 0, stream>>>(x2, bv, gate, gden, attacc);

    // ---- fusion head ----
    k_head<<<dim3(BB), dim3(128), 0, stream>>>(meanacc, maxacc, attacc, gcnt,
        clinical, metab, f_w1, f_b1, f_w2, f_b2, r_w, r_b, outRisk, outGE, outLat);
}
